// TMClauses_55731495632959
// MI455X (gfx1250) — 8-bit matrix operand order repaired and checked by reading, compiled, not run on hardware
//
#include <hip/hip_runtime.h>
#include <stdint.h>

#define NUM_CLASSES 10
#define CLAUSES_PER_CLASS 200
#define M_TOTAL 2000
#define B_BATCH 32
#define D_LITS 2304
#define L_PATCH 900
#define K_STEPS (D_LITS / 64)

typedef __attribute__((ext_vector_type(8))) int v8i;

#define LDA 80
#define LDB 80

__global__ void zero_or_kernel(int* __restrict__ p, int n) {
    int i = blockIdx.x * blockDim.x + threadIdx.x;
    if (i < n) p[i] = 0;
}

__global__ void mask_to_u8_kernel(const int* __restrict__ m, uint8_t* __restrict__ o, int n) {
    int i = blockIdx.x * blockDim.x + threadIdx.x;
    if (i < n) o[i] = (uint8_t)m[i];
}

__global__ __launch_bounds__(256) void lit_transpose_kernel(const float* __restrict__ lit,
                                                            uint8_t* __restrict__ litT) {
    __shared__ uint8_t tile[32][33];
    const int b  = blockIdx.z;
    const int l0 = blockIdx.x * 32;
    const int d0 = blockIdx.y * 32;
    const int tx = threadIdx.x;
    const int ty = threadIdx.y;

    const float* src = lit + (size_t)b * D_LITS * L_PATCH;
#pragma unroll
    for (int i = 0; i < 4; ++i) {
        const int d = d0 + ty + i * 8;
        const int l = l0 + tx;
        uint8_t v = 0;
        if (l < L_PATCH) {
            const float f = __builtin_nontemporal_load(&src[(size_t)d * L_PATCH + l]);
            v = (f < 0.5f) ? (uint8_t)1 : (uint8_t)0;
        }
        tile[ty + i * 8][tx] = v;
    }
    __syncthreads();
    uint8_t* dst = litT + (size_t)b * L_PATCH * D_LITS;
#pragma unroll
    for (int i = 0; i < 4; ++i) {
        const int l = l0 + ty + i * 8;
        const int d = d0 + tx;
        if (l < L_PATCH) dst[(size_t)l * D_LITS + d] = tile[tx][ty + i * 8];
    }
}

__global__ __launch_bounds__(256) void tm_wmma_kernel(const uint8_t* __restrict__ maskU8,
                                                      const uint8_t* __restrict__ litT,
                                                      float* __restrict__ out_map,
                                                      int* __restrict__ or_flags) {
    __shared__ __align__(16) uint8_t ldsA[128 * LDA];
    __shared__ __align__(16) uint8_t ldsB[128 * LDB];

    const int b  = blockIdx.z;
    const int m0 = blockIdx.y * 128;
    const int l0 = blockIdx.x * 128;

    const int tid  = threadIdx.x;
    const int wave = tid >> 5;
    const int lane = tid & 31;

    const int wm = (wave & 3) * 2;
    const int wl = (wave >> 2) * 4;

    v8i acc[2][4] = {};

    const int sRow  = tid >> 1;
    const int sHalf = (tid & 1) * 32;

    const size_t litBase = (size_t)b * L_PATCH * D_LITS;

    for (int kc = 0; kc < K_STEPS; ++kc) {
        const int k0 = kc * 64;
        __syncthreads();
        {
            uint4 v0 = {}, v1 = {};
            const int gm = m0 + sRow;
            if (gm < M_TOTAL) {
                const uint4* p = (const uint4*)(maskU8 + (size_t)gm * D_LITS + k0 + sHalf);
                v0 = p[0]; v1 = p[1];
            }
            uint4* q = (uint4*)(ldsA + sRow * LDA + sHalf);
            q[0] = v0; q[1] = v1;
        }
        {
            uint4 v0 = {}, v1 = {};
            const int gl = l0 + sRow;
            if (gl < L_PATCH) {
                const uint4* p = (const uint4*)(litT + litBase + (size_t)gl * D_LITS + k0 + sHalf);
                v0 = p[0]; v1 = p[1];
            }
            uint4* q = (uint4*)(ldsB + sRow * LDB + sHalf);
            q[0] = v0; q[1] = v1;
        }
        __syncthreads();

        const int ar = lane & 15;
        const int ak = (lane >> 4) * 16;
        const int bn = lane & 15;
        const int bk = (lane >> 4) * 16;

        v8i afrag[2], bfrag[4];
#pragma unroll
        for (int i = 0; i < 2; ++i) {
            const uint8_t* ap = ldsA + ((wm + i) * 16 + ar) * LDA + ak;
            uint64_t d0 = *(const uint64_t*)(ap + 0);
            uint64_t d1 = *(const uint64_t*)(ap + 8);
            uint64_t d2 = *(const uint64_t*)(ap + 32);
            uint64_t d3 = *(const uint64_t*)(ap + 40);
            v8i a;
            a[0] = (int)(uint32_t)d0; a[1] = (int)(d0 >> 32);
            a[2] = (int)(uint32_t)d1; a[3] = (int)(d1 >> 32);
            a[4] = (int)(uint32_t)d2; a[5] = (int)(d2 >> 32);
            a[6] = (int)(uint32_t)d3; a[7] = (int)(d3 >> 32);
            afrag[i] = a;
        }
#pragma unroll
        for (int j = 0; j < 4; ++j) {
            const uint8_t* bp = ldsB + ((wl + j) * 16 + bn) * LDB + bk;
            uint4 x = *(const uint4*)(bp);
            uint4 y = *(const uint4*)(bp + 32);
            v8i bb;
            bb[0] = (int)x.x; bb[1] = (int)x.y; bb[2] = (int)x.z; bb[3] = (int)x.w;
            bb[4] = (int)y.x; bb[5] = (int)y.y; bb[6] = (int)y.z; bb[7] = (int)y.w;
            bfrag[j] = bb;
        }
#pragma unroll
        for (int i = 0; i < 2; ++i)
#pragma unroll
            for (int j = 0; j < 4; ++j)
                acc[i][j] = __builtin_amdgcn_wmma_i32_16x16x64_iu8(
                    false, afrag[i], false, bfrag[j], acc[i][j], false, false);
    }

    const int lcol   = lane & 15;
    const int rowOfs = (lane >> 4) * 8;
#pragma unroll
    for (int i = 0; i < 2; ++i) {
        const int mTileBase = m0 + (wm + i) * 16;
#pragma unroll
        for (int j = 0; j < 4; ++j) {
            const int l = l0 + (wl + j) * 16 + lcol;
            const bool lvalid = (l < L_PATCH);
            const v8i c = acc[i][j];
            float* rowPtr = out_map + ((size_t)b * M_TOTAL + mTileBase + rowOfs) * L_PATCH + l;
#pragma unroll
            for (int r = 0; r < 8; ++r) {
                const int m = mTileBase + rowOfs + r;
                const bool fired = (c[r] == 0);
                if (lvalid && m < M_TOTAL) {
                    __builtin_nontemporal_store(fired ? 1.0f : 0.0f, rowPtr);
                }
                rowPtr += L_PATCH;
                const unsigned msk = __builtin_amdgcn_ballot_w32(fired && lvalid);
                if (lane == 0) {
                    const int mA = mTileBase + r;
                    const int mB = mTileBase + 8 + r;
                    if ((msk & 0xFFFFu) && mA < M_TOTAL) atomicOr(&or_flags[b * M_TOTAL + mA], 1);
                    if ((msk >> 16)     && mB < M_TOTAL) atomicOr(&or_flags[b * M_TOTAL + mB], 1);
                }
            }
        }
    }
}

__global__ void write_or_kernel(const int* __restrict__ f, float* __restrict__ o, int n) {
    int i = blockIdx.x * blockDim.x + threadIdx.x;
    if (i < n) o[i] = f[i] ? 1.0f : 0.0f;
}

__global__ void logits_kernel(const int* __restrict__ f, const float* __restrict__ alpha,
                              float* __restrict__ o) {
    int i = blockIdx.x * blockDim.x + threadIdx.x;
    if (i >= B_BATCH * NUM_CLASSES) return;
    const int b = i / NUM_CLASSES;
    const int c = i % NUM_CLASSES;
    const int base = c * CLAUSES_PER_CLASS;
    float s = 0.0f;
    for (int j = 0; j < CLAUSES_PER_CLASS; ++j) {
        const int m = base + j;
        const float v = f[b * M_TOTAL + m] ? alpha[m] : 0.0f;
        s += (j < CLAUSES_PER_CLASS / 2) ? v : -v;
    }
    o[i] = s;
}

extern "C" void kernel_launch(void* const* d_in, const int* in_sizes, int n_in,
                              void* d_out, int out_size, void* d_ws, size_t ws_size,
                              hipStream_t stream) {
    const float* literals    = (const float*)d_in[0];
    const int*   clause_mask = (const int*)d_in[1];
    const float* alpha       = (const float*)d_in[2];

    float* out        = (float*)d_out;
    float* out_map    = out;
    float* out_or     = out + (size_t)B_BATCH * M_TOTAL * L_PATCH;
    float* out_logits = out_or + B_BATCH * M_TOTAL;

    uint8_t* ws       = (uint8_t*)d_ws;
    uint8_t* litT     = ws;
    const size_t litT_sz = (size_t)B_BATCH * L_PATCH * D_LITS;
    uint8_t* maskU8   = ws + litT_sz;
    const size_t mask_sz = (size_t)M_TOTAL * D_LITS;
    int* or_flags     = (int*)(ws + litT_sz + mask_sz);

    const int nOr = B_BATCH * M_TOTAL;
    zero_or_kernel<<<(nOr + 255) / 256, 256, 0, stream>>>(or_flags, nOr);

    const int nMask = M_TOTAL * D_LITS;
    mask_to_u8_kernel<<<(nMask + 255) / 256, 256, 0, stream>>>(clause_mask, maskU8, nMask);

    dim3 tb(32, 8), tg((L_PATCH + 31) / 32, D_LITS / 32, B_BATCH);
    lit_transpose_kernel<<<tg, tb, 0, stream>>>(literals, litT);

    dim3 gg((L_PATCH + 127) / 128, (M_TOTAL + 127) / 128, B_BATCH);
    tm_wmma_kernel<<<gg, 256, 0, stream>>>(maskU8, litT, out_map, or_flags);

    write_or_kernel<<<(nOr + 255) / 256, 256, 0, stream>>>(or_flags, out_or, nOr);
    logits_kernel<<<2, 256, 0, stream>>>(or_flags, alpha, out_logits);
}
